// RNN_7249904796201
// MI455X (gfx1250) — compile-verified, matrix operand order checked by reading, not run on hardware
//
#include <hip/hip_runtime.h>

#define B_ 512
#define T_ 1024
#define I_ 64
#define H_ 128

typedef __attribute__((ext_vector_type(16))) __bf16 v16bf;
typedef __attribute__((ext_vector_type(8)))  float  v8f;

union FragB {
  __bf16 h[16];
  v16bf  b;
};

// branchless tanh: 1 - 2/(exp2(2*log2e*x)+1); saturates correctly at +-inf
__device__ __forceinline__ float fast_tanh(float x) {
  float e = __builtin_amdgcn_exp2f(x * 2.88539008177793f);  // 2*log2(e)
  return 1.0f - 2.0f * __builtin_amdgcn_rcpf(e + 1.0f);
}

// One block = 16 batch rows, 256 threads = 8 waves; wave w owns h-columns [16w,16w+16).
// Weight B-fragments (bf16) stay resident in VGPRs for the whole 1024-step scan.
__global__ __launch_bounds__(256) void rnn_scan_kernel(
    const float* __restrict__ x,   const float* __restrict__ Wih,
    const float* __restrict__ Whh, const float* __restrict__ bih,
    const float* __restrict__ bhh, const float* __restrict__ fcw,
    const float* __restrict__ fcb, float* __restrict__ out)
{
  __shared__ __bf16 xs[2][16][I_];   // 4 KB : x_t tile, double buffered
  __shared__ __bf16 hs[2][16][H_];   // 8 KB : hidden state, double buffered

  const int tid  = threadIdx.x;
  const int lane = tid & 31;
  const int wv   = tid >> 5;                 // wave id 0..7
  const int b0   = blockIdx.x * 16;

  const int ncol  = (wv << 4) + (lane & 15); // output column of H this lane produces
  const int khalf = (lane >> 4) << 4;        // 0 or 16 (K-half per lane group)
  const int mrow  = lane & 15;               // A-fragment row

  // ---- preload weight B-fragments (bf16) + bias, once ---------------------
  v16bf wih[2];
  #pragma unroll
  for (int kk = 0; kk < 2; ++kk) {
    FragB fr;
    const float* s = Wih + ncol * I_ + kk * 32 + khalf;   // B[k][n] = Wih[n*I+k]
    #pragma unroll
    for (int i = 0; i < 16; ++i) fr.h[i] = (__bf16)s[i];
    wih[kk] = fr.b;
  }
  v16bf whh[4];
  #pragma unroll
  for (int kk = 0; kk < 4; ++kk) {
    FragB fr;
    const float* s = Whh + ncol * H_ + kk * 32 + khalf;   // B[k][n] = Whh[n*H+k]
    #pragma unroll
    for (int i = 0; i < 16; ++i) fr.h[i] = (__bf16)s[i];
    whh[kk] = fr.b;
  }
  const float bias = bih[ncol] + bhh[ncol];

  // ---- init: stage x_0, zero h0 -------------------------------------------
  const int xrow = tid >> 4;                 // 0..15 batch row in tile
  const int xcol = (tid & 15) << 2;          // 0..60 step 4
  const float* xbase = x + (size_t)(b0 + xrow) * (T_ * I_) + xcol;
  {
    float4 v = *(const float4*)xbase;        // x_0
    xs[0][xrow][xcol + 0] = (__bf16)v.x;
    xs[0][xrow][xcol + 1] = (__bf16)v.y;
    xs[0][xrow][xcol + 2] = (__bf16)v.z;
    xs[0][xrow][xcol + 3] = (__bf16)v.w;
    #pragma unroll
    for (int i = 0; i < 8; ++i)
      hs[0][tid >> 4][((tid & 15) << 3) + i] = (__bf16)0.0f;   // h0 = 0
  }
  __syncthreads();

  // ---- sequential scan over T ---------------------------------------------
  for (int t = 0; t < T_; ++t) {
    const int cur = t & 1, nxt = cur ^ 1;

    // prefetch x_{t+1} into registers while WMMAs run
    float4 pf;
    const bool has = (t + 1 < T_);
    if (has) pf = *(const float4*)(xbase + (size_t)(t + 1) * I_);

    // two independent accumulator chains for ILP
    v8f ci = {};   // x_t @ Wih^T   (K = 64)
    v8f ch = {};   // h_t @ Whh^T   (K = 128)
    #pragma unroll
    for (int kk = 0; kk < 2; ++kk) {
      v16bf ax = *(const v16bf*)&xs[cur][mrow][kk * 32 + khalf];
      v16bf ah = *(const v16bf*)&hs[cur][mrow][kk * 32 + khalf];
      ci = __builtin_amdgcn_wmma_f32_16x16x32_bf16(false, ax, false, wih[kk],
                                                   (short)0, ci, false, false);
      ch = __builtin_amdgcn_wmma_f32_16x16x32_bf16(false, ah, false, whh[kk],
                                                   (short)0, ch, false, false);
    }
    #pragma unroll
    for (int kk = 2; kk < 4; ++kk) {
      v16bf ah = *(const v16bf*)&hs[cur][mrow][kk * 32 + khalf];
      ch = __builtin_amdgcn_wmma_f32_16x16x32_bf16(false, ah, false, whh[kk],
                                                   (short)0, ch, false, false);
    }

    // stage prefetched x_{t+1} (hardware bf16 converts)
    if (has) {
      xs[nxt][xrow][xcol + 0] = (__bf16)pf.x;
      xs[nxt][xrow][xcol + 1] = (__bf16)pf.y;
      xs[nxt][xrow][xcol + 2] = (__bf16)pf.z;
      xs[nxt][xrow][xcol + 3] = (__bf16)pf.w;
    }

    // bias + tanh + write new h tile (C layout: VGPR r -> M = r + 8*(lane>=16))
    const int mbase = (lane >> 4) << 3;
    #pragma unroll
    for (int r = 0; r < 8; ++r) {
      float hval = fast_tanh(ci[r] + ch[r] + bias);
      hs[nxt][mbase + r][ncol] = (__bf16)hval;
    }
    __syncthreads();
  }

  // ---- final FC + sigmoid (tiny, once) ------------------------------------
  const int fin = T_ & 1;                    // buffer holding h_T
  if (tid < 16) {
    float acc = fcb[0];
    #pragma unroll 4
    for (int n = 0; n < H_; ++n) acc += (float)hs[fin][tid][n] * fcw[n];
    out[b0 + tid] = __builtin_amdgcn_rcpf(1.0f + __builtin_amdgcn_exp2f(-acc * 1.44269504088896f));
  }
}

extern "C" void kernel_launch(void* const* d_in, const int* in_sizes, int n_in,
                              void* d_out, int out_size, void* d_ws, size_t ws_size,
                              hipStream_t stream) {
  (void)in_sizes; (void)n_in; (void)d_ws; (void)ws_size; (void)out_size;
  const float* x   = (const float*)d_in[0];
  const float* Wih = (const float*)d_in[1];
  const float* Whh = (const float*)d_in[2];
  const float* bih = (const float*)d_in[3];
  const float* bhh = (const float*)d_in[4];
  const float* fcw = (const float*)d_in[5];
  const float* fcb = (const float*)d_in[6];
  float* out = (float*)d_out;

  rnn_scan_kernel<<<dim3(B_ / 16), dim3(256), 0, stream>>>(
      x, Wih, Whh, bih, bhh, fcw, fcb, out);
}
